// Attention_20040317403548
// MI455X (gfx1250) — matrix operand order repaired and checked by reading, not compiled, not run on hardware
//
#include <hip/hip_runtime.h>
#include <hip/hip_bf16.h>

// Problem constants (match reference)
#define B_    2
#define S_LEN 2048
#define DIM_  2048
#define H_    32
#define KVH_  8
#define HD_   64
#define NREP_ 4

typedef __attribute__((ext_vector_type(16))) _Float16 v16h;
typedef __attribute__((ext_vector_type(8)))  _Float16 v8h;
typedef __attribute__((ext_vector_type(8)))  float    v8f;

// ---------------------------------------------------------------------------
// WMMA fragment loaders (wave32, v_wmma_f32_16x16x32_f16 operand layouts)
// A (16M x 32K), row-major source with leading dim `ld` (elements):
//   lane L holds row M=L&15; K-base kb=8*(L>>4); elems 0..7 -> K=kb..kb+7,
//   elems 8..15 -> K=16+kb..16+kb+7.  (ISA 7.12.2 16-bit A table)
// B (32K x 16N), source given transposed (BT row-major [N][K]):
//   lane L holds col N=L&15; kb=8*(L>>4); elems 0..7 -> K=kb..kb+7,
//   elems 8..15 -> K=16+kb..16+kb+7 (same as A).
// ---------------------------------------------------------------------------
static __device__ __forceinline__ v16h load_frag_a(const _Float16* base, int ld) {
    const int lane = threadIdx.x & 31;
    const _Float16* p = base + (size_t)(lane & 15) * ld + ((lane >> 4) << 3);
    v8h lo = *(const v8h*)(p);
    v8h hi = *(const v8h*)(p + 16);
    v16h r;
#pragma unroll
    for (int i = 0; i < 8; ++i) { r[i] = lo[i]; r[i + 8] = hi[i]; }
    return r;
}

static __device__ __forceinline__ v16h load_frag_b(const _Float16* baseT, int ld) {
    const int lane = threadIdx.x & 31;
    const _Float16* p = baseT + (size_t)(lane & 15) * ld + ((lane >> 4) << 3);
    v8h lo = *(const v8h*)(p);
    v8h hi = *(const v8h*)(p + 16);
    v16h r;
#pragma unroll
    for (int i = 0; i < 8; ++i) { r[i] = lo[i]; r[i + 8] = hi[i]; }
    return r;
}

static __device__ __forceinline__ v8f wmma_f16(v16h a, v16h b, v8f c) {
    return __builtin_amdgcn_wmma_f32_16x16x32_f16(false, a, false, b, (short)0, c,
                                                  false, false);
}

// Async global -> LDS copy, 16 bytes per lane (gfx1250, tracked by ASYNCcnt).
// lds_off is a byte offset into the workgroup LDS segment (kernel uses only
// dynamic LDS, so the dynamic segment starts at offset 0).
static __device__ __forceinline__ void async_copy_b128(unsigned lds_off,
                                                       const void* gaddr) {
    asm volatile("global_load_async_to_lds_b128 %0, %1, off"
                 :
                 : "v"(lds_off), "v"((unsigned long long)(uintptr_t)gaddr)
                 : "memory");
}

static __device__ __forceinline__ void wait_asynccnt0() {
    asm volatile("s_wait_asynccnt 0x0" ::: "memory");
}

// ---------------------------------------------------------------------------
// f32 -> f16 conversion / transposed conversion
// ---------------------------------------------------------------------------
__global__ void cvt_f32_f16(const float* __restrict__ in, _Float16* __restrict__ out,
                            size_t n) {
    size_t stride = (size_t)gridDim.x * blockDim.x;
    for (size_t i = (size_t)blockIdx.x * blockDim.x + threadIdx.x; i < n; i += stride)
        out[i] = (_Float16)in[i];
}

// in: [K][N] f32 row-major  ->  out: [N][K] f16 row-major
__global__ void transpose_f32_f16(const float* __restrict__ in,
                                  _Float16* __restrict__ out, int K, int N) {
    size_t total = (size_t)K * N;
    size_t stride = (size_t)gridDim.x * blockDim.x;
    for (size_t i = (size_t)blockIdx.x * blockDim.x + threadIdx.x; i < total; i += stride) {
        size_t k = i / (size_t)N;
        size_t n = i % (size_t)N;
        out[n * (size_t)K + k] = (_Float16)in[i];
    }
}

// ---------------------------------------------------------------------------
// WMMA GEMM: C[M,N] = A[M,K] * B[K,N], with B supplied transposed (BT [N][K]).
// Block = 256 threads = 8 waves arranged 2(M) x 4(N).
// Wave tile = 32M x 64N: 2 A-fragments x 4 B-fragments -> 8 WMMAs per K-step
// from 6 fragment loads (0.75 loads/WMMA).
// Block tile = 64M x 256N.
// PERMUTE: scatter output as [b, head, s, hd] f16 (for Q/K/V).
// OUT_F16: plain row-major f16 out; else row-major f32 out.
// ---------------------------------------------------------------------------
template <bool OUT_F16, bool PERMUTE>
__global__ __launch_bounds__(256) void wmma_gemm(const _Float16* __restrict__ A,
                                                 const _Float16* __restrict__ BT,
                                                 void* __restrict__ Cout, int M, int N,
                                                 int Kd, int Hn) {
    const int wave = threadIdx.x >> 5;
    const int lane = threadIdx.x & 31;
    const int m0 = blockIdx.x * 64 + (wave >> 2) * 32;     // 2 M-rows of waves
    const int n0 = blockIdx.y * 256 + (wave & 3) * 64;     // 4 N-cols of waves

    v8f c[2][4];
#pragma unroll
    for (int i = 0; i < 2; ++i)
#pragma unroll
        for (int j = 0; j < 4; ++j) c[i][j] = v8f{};

    const _Float16* Arow0 = A + (size_t)m0 * Kd;
    const _Float16* Arow1 = A + (size_t)(m0 + 16) * Kd;

    for (int k = 0; k < Kd; k += 32) {
        v16h a0 = load_frag_a(Arow0 + k, Kd);
        v16h a1 = load_frag_a(Arow1 + k, Kd);
#pragma unroll
        for (int j = 0; j < 4; ++j) {
            v16h bj = load_frag_b(BT + (size_t)(n0 + j * 16) * Kd + k, Kd);
            c[0][j] = wmma_f16(a0, bj, c[0][j]);
            c[1][j] = wmma_f16(a1, bj, c[1][j]);
        }
    }

    const int col = lane & 15;
    const int ro  = (lane >> 4) << 3;
#pragma unroll
    for (int i = 0; i < 2; ++i)
#pragma unroll
        for (int j = 0; j < 4; ++j)
#pragma unroll
            for (int v = 0; v < 8; ++v) {
                const int m = m0 + i * 16 + ro + v;
                const int n = n0 + j * 16 + col;
                const float val = c[i][j][v];
                if (PERMUTE) {
                    int b = m / S_LEN, s = m % S_LEN;
                    int h = n / HD_, d = n % HD_;
                    size_t idx = ((size_t)(b * Hn + h) * S_LEN + s) * HD_ + d;
                    ((_Float16*)Cout)[idx] = (_Float16)val;
                } else if (OUT_F16) {
                    ((_Float16*)Cout)[(size_t)m * N + n] = (_Float16)val;
                } else {
                    ((float*)Cout)[(size_t)m * N + n] = val;
                }
            }
}

// ---------------------------------------------------------------------------
// RoPE in place on [B, heads, S, HD] f16
// ---------------------------------------------------------------------------
__global__ void rope_kernel(_Float16* __restrict__ X, const float* __restrict__ cosT,
                            const float* __restrict__ sinT, int heads) {
    size_t total = (size_t)B_ * heads * S_LEN * (HD_ / 2);
    size_t stride = (size_t)gridDim.x * blockDim.x;
    for (size_t i = (size_t)blockIdx.x * blockDim.x + threadIdx.x; i < total; i += stride) {
        int pair = (int)(i & 31);            // HD/2 = 32
        size_t rem = i >> 5;                 // (b*heads + h)*S + s
        int s = (int)(rem & (S_LEN - 1));
        float c = cosT[s * 32 + pair];
        float sn = sinT[s * 32 + pair];
        _Float16* p = X + rem * HD_ + pair * 2;
        float a = (float)p[0], b = (float)p[1];
        p[0] = (_Float16)(a * c - b * sn);
        p[1] = (_Float16)(a * sn + b * c);
    }
}

// ---------------------------------------------------------------------------
// Flash attention: grid (S/64, B*H), block 128 = 4 waves.
// Dynamic-LDS layout (no static __shared__, so dynamic segment is offset 0):
//   [    0 ..  9216) Vt  : V^T tile [d][k], 64 x 72 halves (padded rows)
//   [ 9216 .. 18432) Ps  : per-wave P tiles, 4 x 16 x 72 halves
//   [18432 .. 34816) Raw : double-buffered raw V tiles, 2 x 4096 halves
// Pipeline per key tile:
//   s_wait_asynccnt 0 -> barrier -> issue async copy of NEXT V tile
//   (global_load_async_to_lds_b128) -> LDS->LDS transpose of current tile
//   -> barrier -> WMMA scores -> online softmax -> WMMA O += P*V.
// Output: Ao [B, S, H, HD] f16 (A matrix of the wo GEMM).
// ---------------------------------------------------------------------------
__global__ __launch_bounds__(128) void flash_attn(const _Float16* __restrict__ Qp,
                                                  const _Float16* __restrict__ Kp,
                                                  const _Float16* __restrict__ Vp,
                                                  _Float16* __restrict__ Ao) {
    extern __shared__ char smem[];
    _Float16* Vt = (_Float16*)(smem);                   // 64*72 halves
    _Float16* Ps = (_Float16*)(smem + 9216);            // 4*16*72 halves
    const unsigned RAW_OFF = 18432;                     // byte offset of raw bufs

    const int wave = threadIdx.x >> 5;
    const int lane = threadIdx.x & 31;
    const int qt = blockIdx.x;
    const int bh = (int)blockIdx.y;
    const int b = bh / H_;
    const int h = bh % H_;
    const int kvh = h / NREP_;            // GQA: repeat along head dim
    const int qrow0 = qt * 64 + wave * 16;

    const _Float16* Qb = Qp + ((size_t)(b * H_ + h) * S_LEN + qrow0) * HD_;
    const v16h qa0 = load_frag_a(Qb, HD_);
    const v16h qa1 = load_frag_a(Qb + 32, HD_);

    const _Float16* Kbase = Kp + (size_t)(b * KVH_ + kvh) * S_LEN * HD_;
    const _Float16* Vbase = Vp + (size_t)(b * KVH_ + kvh) * S_LEN * HD_;

    v8f o[4];
    float mrow[8], lrow[8];
#pragma unroll
    for (int nb = 0; nb < 4; ++nb) o[nb] = v8f{};
#pragma unroll
    for (int v = 0; v < 8; ++v) { mrow[v] = -3.0e38f; lrow[v] = 0.0f; }

    // ---- prologue: async-copy V tile 0 into raw buffer 0 ----
    {
        const char* g = (const char*)Vbase;
#pragma unroll
        for (int cch = 0; cch < 4; ++cch) {
            int e = cch * 128 + (int)threadIdx.x;       // 16B chunk index
            async_copy_b128(RAW_OFF + (unsigned)e * 16, g + (size_t)e * 16);
        }
    }

    for (int kt = 0; kt <= qt; ++kt) {
        // ---- current V tile resident in LDS? (per-wave wait + block barrier)
        wait_asynccnt0();
        __syncthreads();

        // ---- kick off async copy of the NEXT V tile (other raw buffer) ----
        if (kt + 1 <= qt) {
            const char* g = (const char*)(Vbase + (size_t)(kt + 1) * 64 * HD_);
            unsigned lb = RAW_OFF + (unsigned)((kt + 1) & 1) * 8192;
#pragma unroll
            for (int cch = 0; cch < 4; ++cch) {
                int e = cch * 128 + (int)threadIdx.x;
                async_copy_b128(lb + (unsigned)e * 16, g + (size_t)e * 16);
            }
            // prefetch next K tile into cache while we're at it
            __builtin_prefetch((const char*)(Kbase + (size_t)(kt + 1) * 64 * HD_) +
                                   (size_t)threadIdx.x * 64, 0, 3);
        }

        // ---- LDS->LDS transpose: raw [k][d] -> Vt [d][k] (padded rows) ----
        const _Float16* raw =
            (const _Float16*)(smem + RAW_OFF + (unsigned)(kt & 1) * 8192);
        for (int t = threadIdx.x; t < (64 * 64) / 8; t += 128) {
            v8h xv = *(const v8h*)(raw + (size_t)t * 8);
            int k = t >> 3;
            int d0 = (t & 7) * 8;
#pragma unroll
            for (int i = 0; i < 8; ++i) Vt[(d0 + i) * 72 + k] = xv[i];
        }
        __syncthreads();

        // ---- scores: 16 x 64 per wave, two K-steps over HD ----
        const _Float16* Kb = Kbase + (size_t)kt * 64 * HD_;
        v8f sc[4];
#pragma unroll
        for (int nb = 0; nb < 4; ++nb) {
            v8f cc = {};
            cc = wmma_f16(qa0, load_frag_b(Kb + (size_t)(nb * 16) * HD_, HD_), cc);
            cc = wmma_f16(qa1, load_frag_b(Kb + (size_t)(nb * 16) * HD_ + 32, HD_), cc);
            sc[nb] = cc;
        }

        // ---- scale + causal mask ----
        const int colb = kt * 64 + (lane & 15);
        const int rowb = qrow0 + ((lane >> 4) << 3);
#pragma unroll
        for (int nb = 0; nb < 4; ++nb)
#pragma unroll
            for (int v = 0; v < 8; ++v) {
                float sv = sc[nb][v] * 0.125f;  // 1/sqrt(64)
                if (colb + nb * 16 > rowb + v) sv = -3.0e38f;
                sc[nb][v] = sv;
            }

        // ---- online softmax (row = v + 8*(lane>>4); cols across 16-lane half)
#pragma unroll
        for (int v = 0; v < 8; ++v) {
            float mx = fmaxf(fmaxf(sc[0][v], sc[1][v]), fmaxf(sc[2][v], sc[3][v]));
            mx = fmaxf(mx, __shfl_xor(mx, 1, 32));
            mx = fmaxf(mx, __shfl_xor(mx, 2, 32));
            mx = fmaxf(mx, __shfl_xor(mx, 4, 32));
            mx = fmaxf(mx, __shfl_xor(mx, 8, 32));
            float mnew = fmaxf(mrow[v], mx);
            float corr = __expf(mrow[v] - mnew);
            float sum = 0.0f;
#pragma unroll
            for (int nb = 0; nb < 4; ++nb) {
                float p = __expf(sc[nb][v] - mnew);
                sc[nb][v] = p;
                sum += p;
            }
            sum += __shfl_xor(sum, 1, 32);
            sum += __shfl_xor(sum, 2, 32);
            sum += __shfl_xor(sum, 4, 32);
            sum += __shfl_xor(sum, 8, 32);
            mrow[v] = mnew;
            lrow[v] = lrow[v] * corr + sum;
#pragma unroll
            for (int nb = 0; nb < 4; ++nb) o[nb][v] *= corr;
        }

        // ---- P (C-layout f32) -> wave-private LDS (row-major f16) ----
        _Float16* Pw = Ps + wave * (16 * 72);
#pragma unroll
        for (int nb = 0; nb < 4; ++nb)
#pragma unroll
            for (int v = 0; v < 8; ++v)
                Pw[(((lane >> 4) << 3) + v) * 72 + nb * 16 + (lane & 15)] =
                    (_Float16)sc[nb][v];
        // LDS ops from one wave execute in order: store->load safe, no barrier.

        const v16h pa0 = load_frag_a(Pw, 72);
        const v16h pa1 = load_frag_a(Pw + 32, 72);
#pragma unroll
        for (int nb = 0; nb < 4; ++nb) {
            o[nb] = wmma_f16(pa0, load_frag_b(&Vt[nb * 16 * 72], 72), o[nb]);
            o[nb] = wmma_f16(pa1, load_frag_b(&Vt[nb * 16 * 72] + 32, 72), o[nb]);
        }
    }

    // ---- epilogue: O /= l, store to Ao[b, s, h, d] f16 ----
    const int col = lane & 15;
    const int ro = (lane >> 4) << 3;
#pragma unroll
    for (int v = 0; v < 8; ++v) {
        int r = qrow0 + ro + v;
        float inv = 1.0f / lrow[v];
        size_t base = ((size_t)(b * S_LEN + r) * H_ + h) * HD_;
#pragma unroll
        for (int nb = 0; nb < 4; ++nb)
            Ao[base + nb * 16 + col] = (_Float16)(o[nb][v] * inv);
    }
}

// ---------------------------------------------------------------------------
// Launcher
// Inputs: x, wq, wk, wv, wo, freqs_cos, freqs_sin, mask(unused), start_pos(unused)
// ---------------------------------------------------------------------------
extern "C" void kernel_launch(void* const* d_in, const int* in_sizes, int n_in,
                              void* d_out, int out_size, void* d_ws, size_t ws_size,
                              hipStream_t stream) {
    (void)in_sizes; (void)n_in; (void)out_size; (void)ws_size;
    const float* x  = (const float*)d_in[0];
    const float* wq = (const float*)d_in[1];
    const float* wk = (const float*)d_in[2];
    const float* wv = (const float*)d_in[3];
    const float* wo = (const float*)d_in[4];
    const float* fc = (const float*)d_in[5];
    const float* fs = (const float*)d_in[6];

    char* ws = (char*)d_ws;
    size_t off = 0;
    auto alloc16 = [&](size_t elems) -> _Float16* {
        _Float16* p = (_Float16*)(ws + off);
        off += (elems * sizeof(_Float16) + 255) & ~(size_t)255;
        return p;
    };
    _Float16* xh  = alloc16((size_t)B_ * S_LEN * DIM_);
    _Float16* wqT = alloc16((size_t)DIM_ * DIM_);
    _Float16* wkT = alloc16((size_t)DIM_ * KVH_ * HD_);
    _Float16* wvT = alloc16((size_t)DIM_ * KVH_ * HD_);
    _Float16* woT = alloc16((size_t)DIM_ * DIM_);
    _Float16* Qp  = alloc16((size_t)B_ * H_ * S_LEN * HD_);
    _Float16* Kpp = alloc16((size_t)B_ * KVH_ * S_LEN * HD_);
    _Float16* Vpp = alloc16((size_t)B_ * KVH_ * S_LEN * HD_);
    _Float16* Ao  = alloc16((size_t)B_ * S_LEN * H_ * HD_);

    // 1) convert activations + transpose-convert weights to f16
    cvt_f32_f16<<<2048, 256, 0, stream>>>(x, xh, (size_t)B_ * S_LEN * DIM_);
    transpose_f32_f16<<<2048, 256, 0, stream>>>(wq, wqT, DIM_, H_ * HD_);
    transpose_f32_f16<<<1024, 256, 0, stream>>>(wk, wkT, DIM_, KVH_ * HD_);
    transpose_f32_f16<<<1024, 256, 0, stream>>>(wv, wvT, DIM_, KVH_ * HD_);
    transpose_f32_f16<<<2048, 256, 0, stream>>>(wo, woT, H_ * HD_, DIM_);

    // 2) Q/K/V projections (WMMA), permuted to [B, heads, S, HD]
    wmma_gemm<true, true><<<dim3(64, 8), 256, 0, stream>>>(
        xh, wqT, Qp, B_ * S_LEN, H_ * HD_, DIM_, H_);
    wmma_gemm<true, true><<<dim3(64, 2), 256, 0, stream>>>(
        xh, wkT, Kpp, B_ * S_LEN, KVH_ * HD_, DIM_, KVH_);
    wmma_gemm<true, true><<<dim3(64, 2), 256, 0, stream>>>(
        xh, wvT, Vpp, B_ * S_LEN, KVH_ * HD_, DIM_, KVH_);

    // 3) RoPE on Q and K
    rope_kernel<<<1024, 256, 0, stream>>>(Qp, fc, fs, H_);
    rope_kernel<<<512, 256, 0, stream>>>(Kpp, fc, fs, KVH_);

    // 4) causal flash attention (WMMA + async global->LDS double buffering)
    flash_attn<<<dim3(S_LEN / 64, B_ * H_), 128, 34816, stream>>>(Qp, Kpp, Vpp, Ao);

    // 5) output projection (WMMA), f32 result straight to d_out
    wmma_gemm<false, false><<<dim3(64, 8), 256, 0, stream>>>(
        Ao, woT, d_out, B_ * S_LEN, DIM_, DIM_, 0);
}
